// GlobAttRNNDecoder_18622978195959
// MI455X (gfx1250) — compile-verified, matrix operand order checked by reading, not run on hardware
//
#include <hip/hip_runtime.h>

// GlobAttRNNDecoder for MI455X (gfx1250, wave32).
// Bottleneck: enc_h streaming (scores+ctx passes) -> 65.5 GB bf16 -> ~2.8 ms @ 23.3 TB/s.
// GEMMs: 188 GFLOP total on v_wmma_f32_16x16x32_bf16 (fp32 accumulate) -> latency-bound,
// so the GEMM kernel is branch-free in the K-loop (templated DUAL), software-pipelined
// (prefetch next K-step during WMMA), and runs two independent accumulator chains/wave.

typedef unsigned short u16;
typedef unsigned int   u32;
typedef __attribute__((ext_vector_type(16))) __bf16 v16bf;
typedef __attribute__((ext_vector_type(8)))  float  v8f;

union FragB16 { uint4 q[2]; v16bf v; };

static constexpr int Bc = 64;    // batch
static constexpr int Tc = 1024;  // encoder length
static constexpr int Hc = 1024;  // hidden
static constexpr int Sc = 128;   // decode steps
static constexpr int GK = 1024;  // GEMM K

// ---------- helpers ----------
__device__ __forceinline__ u16 f2bf(float x) {
  u32 u = __float_as_uint(x);
  u32 r = u + 0x7fffu + ((u >> 16) & 1u);   // round-to-nearest-even
  return (u16)(r >> 16);
}
__device__ __forceinline__ u32 pack2(float a, float b) {
  return (u32)f2bf(a) | ((u32)f2bf(b) << 16);
}
__device__ __forceinline__ float bf_lo(u32 u) { return __uint_as_float(u << 16); }
__device__ __forceinline__ float bf_hi(u32 u) { return __uint_as_float(u & 0xffff0000u); }

__device__ __forceinline__ float warp_max(float v) {
  for (int o = 16; o; o >>= 1) v = fmaxf(v, __shfl_down(v, o));
  return v;
}
__device__ __forceinline__ float warp_sum(float v) {
  for (int o = 16; o; o >>= 1) v += __shfl_down(v, o);
  return v;
}
__device__ __forceinline__ float block_max(float v, float* sm) {
  v = warp_max(v);
  if ((threadIdx.x & 31) == 0) sm[threadIdx.x >> 5] = v;
  __syncthreads();
  float r = sm[0];
  #pragma unroll
  for (int i = 1; i < 8; i++) r = fmaxf(r, sm[i]);
  __syncthreads();
  return r;
}
__device__ __forceinline__ float block_sum(float v, float* sm) {
  v = warp_sum(v);
  if ((threadIdx.x & 31) == 0) sm[threadIdx.x >> 5] = v;
  __syncthreads();
  float r = sm[0];
  #pragma unroll
  for (int i = 1; i < 8; i++) r += sm[i];
  __syncthreads();
  return r;
}

// ---------- fp32 -> packed bf16x2 conversion ----------
__global__ __launch_bounds__(256) void k_cvt(const float2* __restrict__ in,
                                             u32* __restrict__ out, int pairs) {
  int stride = gridDim.x * blockDim.x;
  for (int i = blockIdx.x * blockDim.x + threadIdx.x; i < pairs; i += stride) {
    float2 f = in[i];
    out[i] = pack2(f.x, f.y);
  }
}

// ---------- embedding gather -> bf16 input ----------
__global__ __launch_bounds__(256) void k_emb(const int* __restrict__ target,
                                             const float* __restrict__ emb,
                                             u32* __restrict__ inp, int col) {
  int b = blockIdx.x;
  int idx = target[b * Sc + col];
  const float2* e = (const float2*)(emb + (size_t)idx * Hc);
  for (int i = threadIdx.x; i < Hc / 2; i += 256) {
    float2 f = e[i];
    inp[b * (Hc / 2) + i] = pack2(f.x, f.y);
  }
}

// ---------- WMMA GEMM: Y(64xN) = act(Xa·Waᵀ [+ Xb·Wbᵀ] + bias0 [+ bias1]) ----------
// bf16 in, fp32 accumulate. Block = 256 thr = 8 waves; wave computes a 16x32 tile
// (one shared A fragment, two independent accumulator chains). grid.x = N/64.
// K-loop is branch-free (DUAL templated) and software-pipelined: fragments for step
// i+1 are issued before the WMMAs of step i, so s_wait_loadcnt overlaps XDL work.
template <bool DUAL>
__global__ __launch_bounds__(256) void k_gemm(
    const u16* __restrict__ Xa, const u16* __restrict__ Wa,
    const u16* __restrict__ Xb, const u16* __restrict__ Wb,
    const float* __restrict__ bias0, const float* __restrict__ bias1,
    float* __restrict__ Yf, u16* __restrict__ Ybf, int N, int act) {
  const int lane = threadIdx.x & 31;
  const int wave = threadIdx.x >> 5;
  const int mt = wave & 3;           // M tile (M = 64 = 4x16)
  const int np = wave >> 2;          // N pair (0..1): 32 cols each
  const int hf = lane >> 4;          // lane half (ISA 16-bit A/B layout)
  const int l16 = lane & 15;
  const int ng0 = blockIdx.x * 64 + np * 32 + l16;  // output col / W row, tile 0
  const int ng1 = ng0 + 16;                          // tile 1
  const int mrow = mt * 16 + l16;                    // A row

  // ISA 16-bit 16x32 fragment: per lane two 16B chunks at K = 8*half and 16+8*half.
  const u16* pa  = Xa + (size_t)mrow * GK + 8 * hf;
  const u16* pw0 = Wa + (size_t)ng0  * GK + 8 * hf;
  const u16* pw1 = Wa + (size_t)ng1  * GK + 8 * hf;
  const u16* pb  = DUAL ? (Xb + (size_t)mrow * GK + 8 * hf) : (const u16*)0;
  const u16* pv0 = DUAL ? (Wb + (size_t)ng0  * GK + 8 * hf) : (const u16*)0;
  const u16* pv1 = DUAL ? (Wb + (size_t)ng1  * GK + 8 * hf) : (const u16*)0;

  auto LD = [](FragB16& f, const u16* p, int kk) {
    f.q[0] = *(const uint4*)(p + kk);
    f.q[1] = *(const uint4*)(p + kk + 16);
  };

  FragB16 a, w0, w1, b, v0, v1;
  LD(a, pa, 0); LD(w0, pw0, 0); LD(w1, pw1, 0);
  if (DUAL) { LD(b, pb, 0); LD(v0, pv0, 0); LD(v1, pv1, 0); }

  v8f acc0 = {}, acc1 = {};
  constexpr int KI = GK / 32;
  #pragma unroll
  for (int i = 0; i < KI; i++) {
    FragB16 an, w0n, w1n, bn, v0n, v1n;
    if (i + 1 < KI) {                       // prefetch next K-step (uniform, unrolled)
      const int kk = (i + 1) * 32;
      LD(an, pa, kk); LD(w0n, pw0, kk); LD(w1n, pw1, kk);
      if (DUAL) { LD(bn, pb, kk); LD(v0n, pv0, kk); LD(v1n, pv1, kk); }
    }
    acc0 = __builtin_amdgcn_wmma_f32_16x16x32_bf16(false, a.v, false, w0.v,
                                                   (short)0, acc0, false, false);
    acc1 = __builtin_amdgcn_wmma_f32_16x16x32_bf16(false, a.v, false, w1.v,
                                                   (short)0, acc1, false, false);
    if (DUAL) {
      acc0 = __builtin_amdgcn_wmma_f32_16x16x32_bf16(false, b.v, false, v0.v,
                                                     (short)0, acc0, false, false);
      acc1 = __builtin_amdgcn_wmma_f32_16x16x32_bf16(false, b.v, false, v1.v,
                                                     (short)0, acc1, false, false);
    }
    a = an; w0 = w0n; w1 = w1n;
    if (DUAL) { b = bn; v0 = v0n; v1 = v1n; }
  }

  float bs0 = (bias0 ? bias0[ng0] : 0.f) + (bias1 ? bias1[ng0] : 0.f);
  float bs1 = (bias0 ? bias0[ng1] : 0.f) + (bias1 ? bias1[ng1] : 0.f);
  #pragma unroll
  for (int r = 0; r < 8; r++) {
    int mg = mt * 16 + hf * 8 + r;   // C/D layout: VGPR r -> M = r + 8*half
    float x0 = acc0[r] + bs0;
    float x1 = acc1[r] + bs1;
    if (act) { x0 = tanhf(x0); x1 = tanhf(x1); }
    if (Yf)  { Yf[(size_t)mg * N + ng0] = x0; Yf[(size_t)mg * N + ng1] = x1; }
    if (Ybf) { Ybf[(size_t)mg * N + ng0] = f2bf(x0); Ybf[(size_t)mg * N + ng1] = f2bf(x1); }
  }
}

// ---------- attention scores: scores[b,t] = q[b]·enc[b,t] (bf16 data, fp32 acc) ----------
__global__ __launch_bounds__(256) void k_scores(const u32* __restrict__ q,
                                                const u32* __restrict__ enc,
                                                const unsigned char* __restrict__ mask,
                                                float* __restrict__ sc) {
  int lane = threadIdx.x & 31;
  int row = blockIdx.x * 8 + (threadIdx.x >> 5);  // row in [0, B*T)
  int b = row >> 10;
  const u32* er = enc + (size_t)row * (Hc / 2);
  const u32* qr = q + (size_t)b * (Hc / 2);
  float s = 0.f;
  #pragma unroll
  for (int i = 0; i < (Hc / 2) / 32; i++) {
    u32 ue = er[i * 32 + lane];
    u32 uq = qr[i * 32 + lane];
    s = fmaf(bf_lo(ue), bf_lo(uq), s);
    s = fmaf(bf_hi(ue), bf_hi(uq), s);
  }
  s = warp_sum(s);
  if (lane == 0) sc[row] = mask[row] ? s : -1.0e12f;
}

// ---------- softmax over T=1024, in place, one block per batch row ----------
__global__ __launch_bounds__(256) void k_softmax(float* __restrict__ sc) {
  __shared__ float sm[8];
  float* p = sc + (size_t)blockIdx.x * Tc;
  float v[4];
  float m = -3.0e38f;
  #pragma unroll
  for (int i = 0; i < 4; i++) { v[i] = p[threadIdx.x + i * 256]; m = fmaxf(m, v[i]); }
  m = block_max(m, sm);
  float s = 0.f;
  #pragma unroll
  for (int i = 0; i < 4; i++) { v[i] = __expf(v[i] - m); s += v[i]; }
  s = block_sum(s, sm);
  float inv = 1.0f / s;
  #pragma unroll
  for (int i = 0; i < 4; i++) p[threadIdx.x + i * 256] = v[i] * inv;
}

// ---------- context: h[b,:] = sum_t att[b,t] * enc[b,t,:] -> bf16 state ----------
__global__ __launch_bounds__(256) void k_ctx(const float* __restrict__ att,
                                             const u32* __restrict__ enc,
                                             u32* __restrict__ hout) {
  __shared__ float s_att[Tc];
  int b = blockIdx.x;
  for (int i = threadIdx.x; i < Tc; i += 256) s_att[i] = att[(size_t)b * Tc + i];
  __syncthreads();
  int col = blockIdx.y * 256 + threadIdx.x;   // packed index in [0, 512)
  const u32* eb = enc + (size_t)b * Tc * (Hc / 2) + col;
  float a0 = 0.f, a1 = 0.f;
  #pragma unroll 4
  for (int t = 0; t < Tc; t++) {
    float w = s_att[t];
    u32 u = eb[(size_t)t * (Hc / 2)];
    a0 = fmaf(bf_lo(u), w, a0);
    a1 = fmaf(bf_hi(u), w, a1);
  }
  hout[(size_t)b * (Hc / 2) + col] = pack2(a0, a1);
}

// ---------- log_softmax over C=1024 -> d_out[b, s, :] ----------
__global__ __launch_bounds__(256) void k_lsm(const float* __restrict__ lin,
                                             float* __restrict__ out, int s) {
  __shared__ float sm[8];
  const float* p = lin + (size_t)blockIdx.x * Hc;
  float v[4];
  float m = -3.0e38f;
  #pragma unroll
  for (int i = 0; i < 4; i++) { v[i] = p[threadIdx.x + i * 256]; m = fmaxf(m, v[i]); }
  m = block_max(m, sm);
  float sum = 0.f;
  #pragma unroll
  for (int i = 0; i < 4; i++) sum += __expf(v[i] - m);
  sum = block_sum(sum, sm);
  float lse = m + __logf(sum);
  float* o = out + ((size_t)blockIdx.x * Sc + s) * Hc;
  #pragma unroll
  for (int i = 0; i < 4; i++) o[threadIdx.x + i * 256] = v[i] - lse;
}

// ---------- host ----------
extern "C" void kernel_launch(void* const* d_in, const int* in_sizes, int n_in,
                              void* d_out, int out_size, void* d_ws, size_t ws_size,
                              hipStream_t stream) {
  const float* h0     = (const float*)d_in[0];               // (1,B,H)
  const float* enc    = (const float*)d_in[1];               // (B,T,H)
  const unsigned char* mask = (const unsigned char*)d_in[2]; // (B,T) bool
  const int*   target = (const int*)d_in[3];                 // (B,128)
  const float* emb    = (const float*)d_in[4];               // (H,E)
  const float* W_ih   = (const float*)d_in[5];               // (2,H,H)
  const float* W_hh   = (const float*)d_in[6];
  const float* b_ih   = (const float*)d_in[7];               // (2,H)
  const float* b_hh   = (const float*)d_in[8];
  const float* Wq     = (const float*)d_in[9];
  const float* bq     = (const float*)d_in[10];
  const float* Wp     = (const float*)d_in[11];              // (C,H)
  const float* bp     = (const float*)d_in[12];
  float* out = (float*)d_out;

  // workspace carve-up (~143 MB)
  char* w = (char*)d_ws;
  auto carve = [&](size_t bytes) -> char* {
    char* p = w; w += (bytes + 255) & ~(size_t)255; return p;
  };
  u16* enc_bf = (u16*)carve((size_t)Bc * Tc * Hc * 2);
  u16* wih_bf = (u16*)carve((size_t)2 * Hc * Hc * 2);
  u16* whh_bf = (u16*)carve((size_t)2 * Hc * Hc * 2);
  u16* wq_bf  = (u16*)carve((size_t)2 * Hc * Hc * 2);
  u16* wp_bf  = (u16*)carve((size_t)Hc * Hc * 2);
  u16* h_bf   = (u16*)carve((size_t)Bc * Hc * 2);   // attention-context state
  u16* inp_bf = (u16*)carve((size_t)Bc * Hc * 2);   // embedded token
  u16* out0_bf = (u16*)carve((size_t)Bc * Hc * 2);  // tanh out, layer 0
  u16* out1_bf = (u16*)carve((size_t)Bc * Hc * 2);  // tanh out, layer 1
  u16* q_bf    = (u16*)carve((size_t)Bc * Hc * 2);
  float* scores = (float*)carve((size_t)Bc * Tc * 4);
  float* lin    = (float*)carve((size_t)Bc * Hc * 4);

  auto cvt = [&](const float* src, u16* dst, size_t pairs) {
    int g = (int)((pairs + 1023) / 1024);   // 4 pairs/thread via grid-stride
    k_cvt<<<g, 256, 0, stream>>>((const float2*)src, (u32*)dst, (int)pairs);
  };
  // one-time conversions (re-run each call; deterministic)
  cvt(enc,  enc_bf, (size_t)Bc * Tc * Hc / 2);
  cvt(W_ih, wih_bf, (size_t)2 * Hc * Hc / 2);
  cvt(W_hh, whh_bf, (size_t)2 * Hc * Hc / 2);
  cvt(Wq,   wq_bf,  (size_t)2 * Hc * Hc / 2);
  cvt(Wp,   wp_bf,  (size_t)Hc * Hc / 2);
  cvt(h0,   h_bf,   (size_t)Bc * Hc / 2);   // initial hidden state

  u16* outl[2] = { out0_bf, out1_bf };

  for (int s = 0; s < Sc; ++s) {
    k_emb<<<Bc, 256, 0, stream>>>(target, emb, (u32*)inp_bf, s == 0 ? 0 : s - 1);
    const u16* X = inp_bf;
    for (int l = 0; l < 2; ++l) {
      const size_t wo = (size_t)l * Hc * Hc;
      // h = tanh(X·W_ihᵀ + h·W_hhᵀ + b_ih + b_hh)   (fused 2-input WMMA GEMM)
      k_gemm<true><<<Hc / 64, 256, 0, stream>>>(X, wih_bf + wo, h_bf, whh_bf + wo,
                                                b_ih + l * Hc, b_hh + l * Hc,
                                                (float*)0, outl[l], Hc, 1);
      // q = h·Wqᵀ + bq
      k_gemm<false><<<Hc / 64, 256, 0, stream>>>(outl[l], wq_bf + wo,
                                                 (const u16*)0, (const u16*)0,
                                                 bq + l * Hc, (const float*)0,
                                                 (float*)0, q_bf, Hc, 0);
      // attention over enc_h (bandwidth-bound path, bf16 streams)
      k_scores<<<(Bc * Tc) / 8, 256, 0, stream>>>((const u32*)q_bf, (const u32*)enc_bf,
                                                  mask, scores);
      k_softmax<<<Bc, 256, 0, stream>>>(scores);
      k_ctx<<<dim3(Bc, 2), 256, 0, stream>>>(scores, (const u32*)enc_bf, (u32*)h_bf);
      X = outl[l];
    }
    // logits = log_softmax(out·Wpᵀ + bp)
    k_gemm<false><<<Hc / 64, 256, 0, stream>>>(outl[1], wp_bf,
                                               (const u16*)0, (const u16*)0,
                                               bp, (const float*)0, lin, (u16*)0, Hc, 0);
    k_lsm<<<Bc, 256, 0, stream>>>(lin, out, s);
  }
}
